// Conv1dLoRA_37709812859091
// MI455X (gfx1250) — matrix operand order repaired and checked by reading, not compiled, not run on hardware
//
#include <hip/hip_runtime.h>
#include <math.h>

typedef __attribute__((ext_vector_type(16))) _Float16 v16h;
typedef __attribute__((ext_vector_type(8)))  float    v8f;

#define BS     128
#define CIN    64
#define COUT   64
#define LLEN   8192
#define KW     3
#define RR     8
#define EMB    256
#define HID    256
#define KDIM   192      // CIN*KW reduction length
#define LCHUNK 256      // L positions per block (8 waves x 32)

__device__ __forceinline__ float gelu_exact(float x) {
    return 0.5f * x * (1.0f + erff(x * 0.70710678118654752f));
}

// ---------------------------------------------------------------------------
// Kernel 1: per-sample MLPs + rank-8 fusion -> fused f16 conv weight
// Wg layout: [b][co][kk*64 + ci]  (f16), so WMMA K index = kk*64+ci
// ---------------------------------------------------------------------------
__global__ __launch_bounds__(256)
void lora_weight_kernel(const float* __restrict__ aEmb, const float* __restrict__ bEmb,
                        const float* __restrict__ Aw1, const float* __restrict__ Ab1,
                        const float* __restrict__ Ag,  const float* __restrict__ Abeta,
                        const float* __restrict__ Aw2, const float* __restrict__ Ab2,
                        const float* __restrict__ Bw1, const float* __restrict__ Bb1,
                        const float* __restrict__ Bg,  const float* __restrict__ Bbeta,
                        const float* __restrict__ Bw2, const float* __restrict__ Bb2,
                        const float* __restrict__ baseW,
                        _Float16* __restrict__ Wg)
{
    __shared__ float sE[EMB];
    __shared__ float sH[HID];
    __shared__ float sA[CIN * RR];        // 512:  A[ci][r]
    __shared__ float sB[RR * COUT * KW];  // 1536: B[r][co*3+kk]

    const int b = blockIdx.x;
    const int t = threadIdx.x;
    const float bnscale = rsqrtf(1.0f + 1e-5f);

    // ---- MLP A ----
    sE[t] = aEmb[(size_t)b * EMB + t];
    __syncthreads();
    {
        float acc = Ab1[t];
        const float* wrow = Aw1 + (size_t)t * EMB;
        for (int e = 0; e < EMB; ++e) acc = fmaf(sE[e], wrow[e], acc);
        acc = acc * (Ag[t] * bnscale) + Abeta[t];
        sH[t] = gelu_exact(acc);
    }
    __syncthreads();
    for (int j = t; j < CIN * RR; j += 256) {
        float acc = Ab2[j];
        const float* wrow = Aw2 + (size_t)j * HID;
        for (int e = 0; e < HID; ++e) acc = fmaf(sH[e], wrow[e], acc);
        sA[j] = acc;
    }
    __syncthreads();

    // ---- MLP B ----
    sE[t] = bEmb[(size_t)b * EMB + t];
    __syncthreads();
    {
        float acc = Bb1[t];
        const float* wrow = Bw1 + (size_t)t * EMB;
        for (int e = 0; e < EMB; ++e) acc = fmaf(sE[e], wrow[e], acc);
        acc = acc * (Bg[t] * bnscale) + Bbeta[t];
        sH[t] = gelu_exact(acc);
    }
    __syncthreads();
    for (int j = t; j < RR * COUT * KW; j += 256) {
        float acc = Bb2[j];
        const float* wrow = Bw2 + (size_t)j * HID;
        for (int e = 0; e < HID; ++e) acc = fmaf(sH[e], wrow[e], acc);
        sB[j] = acc;
    }
    __syncthreads();

    // ---- Wtot[co][kk*64+ci] = sum_r A[ci][r]*B[r][co*3+kk] + base_w[co][ci][kk]
    _Float16* wout = Wg + (size_t)b * COUT * KDIM;
    for (int idx = t; idx < COUT * KDIM; idx += 256) {
        const int co  = idx / KDIM;
        const int rem = idx - co * KDIM;
        const int kk  = rem / CIN;
        const int ci  = rem - kk * CIN;
        float s = baseW[((size_t)co * CIN + ci) * KW + kk];
        #pragma unroll
        for (int r = 0; r < RR; ++r)
            s = fmaf(sA[ci * RR + r], sB[r * (COUT * KW) + co * KW + kk], s);
        wout[co * KDIM + kk * CIN + ci] = (_Float16)s;
    }
}

// ---------------------------------------------------------------------------
// Kernel 2: per-sample im2col GEMM via WMMA f16 -> f32
// block = 8 waves, covers (sample b, 256 L positions); wave = 64(Cout) x 32(L)
// ---------------------------------------------------------------------------
union FragU { v16h h; unsigned int u[8]; };

__global__ __launch_bounds__(256)
void lora_conv_kernel(const float* __restrict__ X,
                      const _Float16* __restrict__ Wg,
                      const float* __restrict__ baseB,
                      float* __restrict__ out)
{
    __shared__ _Float16 Xs[(LCHUNK + 2) * CIN];  // [li][ci], li = l - (l0-1)
    __shared__ _Float16 Ws[COUT * KDIM];         // [co][kk*64+ci]

    const int  t  = threadIdx.x;
    const int  b  = blockIdx.y;
    const long l0 = (long)blockIdx.x * LCHUNK;

    // stage fused weight (raw f16 bytes, dword copies)
    {
        const unsigned int* src = (const unsigned int*)(Wg + (size_t)b * COUT * KDIM);
        unsigned int* dst = (unsigned int*)Ws;
        for (int i = t; i < COUT * KDIM / 2; i += 256) dst[i] = src[i];
    }
    // stage X tile with +/-1 halo, f32 -> f16, transposed to [l][ci]
    {
        const float* Xb = X + (size_t)b * CIN * LLEN;
        for (int idx = t; idx < CIN * (LCHUNK + 2); idx += 256) {
            const int ci = idx / (LCHUNK + 2);
            const int li = idx - ci * (LCHUNK + 2);
            const long l = l0 - 1 + li;
            float v = (l >= 0 && l < LLEN) ? Xb[(size_t)ci * LLEN + l] : 0.0f;
            Xs[li * CIN + ci] = (_Float16)v;
        }
    }
    __syncthreads();

    const int lane = t & 31;
    const int wave = t >> 5;
    const int ln15 = lane & 15;
    const int hi   = lane >> 4;   // 0 for lanes 0-15, 1 for lanes 16-31
    const int p0   = wave * 32;   // wave's local L offset within chunk

    const v8f vzero = {};
    v8f acc[4][2];
    #pragma unroll
    for (int mt = 0; mt < 4; ++mt)
        #pragma unroll
        for (int nt = 0; nt < 2; ++nt) acc[mt][nt] = vzero;

    #pragma unroll
    for (int ks = 0; ks < 6; ++ks) {          // K_global = ks*32 .. ks*32+31
        const int kk     = ks >> 1;           // conv tap 0..2
        const int cibase = (ks & 1) * 32 + hi * 8;

        // B fragments: B(K,n) = Xs[n+kk][ci], ci = cibase + (j>=4)*16 + (j&3)*2 (+e)
        FragU bf[2];
        #pragma unroll
        for (int nt = 0; nt < 2; ++nt) {
            const int row = p0 + nt * 16 + ln15 + kk;
            #pragma unroll
            for (int j = 0; j < 8; ++j)
                bf[nt].u[j] = *(const unsigned int*)&Xs[row * CIN + cibase + ((j & 4) ? 16 : 0) + (j & 3) * 2];
        }

        #pragma unroll
        for (int mt = 0; mt < 4; ++mt) {
            // A fragment: A(m,K) layout K = hi*8 + (j>=4)*16 + (j&3)*2 (+e)
            FragU af;
            const int m  = mt * 16 + ln15;
            const int kb = ks * 32 + hi * 8;
            #pragma unroll
            for (int j = 0; j < 8; ++j) {
                const int K = kb + ((j & 4) ? 16 : 0) + (j & 3) * 2;
                af.u[j] = *(const unsigned int*)&Ws[m * KDIM + K];
            }
            #pragma unroll
            for (int nt = 0; nt < 2; ++nt) {
                acc[mt][nt] = __builtin_amdgcn_wmma_f32_16x16x32_f16(
                    false, af.h, false, bf[nt].h, (short)0, acc[mt][nt], false, false);
            }
        }
    }

    // epilogue: + base_b, write f32.  C/D layout: M = hi*8 + v, N = ln15
    float* ob = out + (size_t)b * COUT * LLEN;
    #pragma unroll
    for (int mt = 0; mt < 4; ++mt) {
        #pragma unroll
        for (int nt = 0; nt < 2; ++nt) {
            const long l = l0 + p0 + nt * 16 + ln15;
            #pragma unroll
            for (int v = 0; v < 8; ++v) {
                const int co = mt * 16 + hi * 8 + v;
                ob[(size_t)co * LLEN + l] = acc[mt][nt][v] + baseB[co];
            }
        }
    }
}

// ---------------------------------------------------------------------------
extern "C" void kernel_launch(void* const* d_in, const int* in_sizes, int n_in,
                              void* d_out, int out_size, void* d_ws, size_t ws_size,
                              hipStream_t stream) {
    const float* X     = (const float*)d_in[0];
    const float* aEmb  = (const float*)d_in[1];
    const float* bEmb  = (const float*)d_in[2];
    const float* Aw1   = (const float*)d_in[3];
    const float* Ab1   = (const float*)d_in[4];
    const float* Ag    = (const float*)d_in[5];
    const float* Abeta = (const float*)d_in[6];
    const float* Aw2   = (const float*)d_in[7];
    const float* Ab2   = (const float*)d_in[8];
    const float* Bw1   = (const float*)d_in[9];
    const float* Bb1   = (const float*)d_in[10];
    const float* Bg    = (const float*)d_in[11];
    const float* Bbeta = (const float*)d_in[12];
    const float* Bw2   = (const float*)d_in[13];
    const float* Bb2   = (const float*)d_in[14];
    const float* baseW = (const float*)d_in[15];
    const float* baseB = (const float*)d_in[16];
    float* out = (float*)d_out;

    _Float16* Wg = (_Float16*)d_ws;   // BS*COUT*KDIM f16 = 3 MB

    lora_weight_kernel<<<BS, 256, 0, stream>>>(
        aEmb, bEmb, Aw1, Ab1, Ag, Abeta, Aw2, Ab2,
        Bw1, Bb1, Bg, Bbeta, Bw2, Bb2, baseW, Wg);

    dim3 grid(LLEN / LCHUNK, BS);
    lora_conv_kernel<<<grid, 256, 0, stream>>>(X, Wg, baseB, out);
}
